// ChannelCompressor_65017214926933
// MI455X (gfx1250) — matrix operand order repaired and checked by reading, not compiled, not run on hardware
//
#include <hip/hip_runtime.h>

typedef __attribute__((ext_vector_type(16))) _Float16     v16h;
typedef __attribute__((ext_vector_type(8)))  float        v8f;
typedef __attribute__((ext_vector_type(8)))  unsigned int v8u;

namespace {
constexpr int   kBatch = 16;
constexpr int   kC     = 64;      // channels
constexpr int   kP     = 65536;   // H*W
constexpr int   kO     = 6;       // output channels
constexpr float kEps   = 1e-6f;
constexpr int   kTPW   = 8;       // 16-pixel tiles per wave
constexpr int   kBlk1  = 256;     // blocks/batch in stats kernel
constexpr int   kBlk2  = 64;      // blocks/batch in fused kernels (64 blk * 8 waves * 8 tiles * 16 px = 65536)
}

// Channel index held by A-fragment element j for a lane with given khalf (=lane>>4),
// per the CDNA5 16-bit A-matrix 16x32 layout (CDNA5 ISA §7.12.2).
__device__ __forceinline__ int a_chan(int j, int khalf) {
  int v = j >> 1, hh = j & 1;
  return ((v & 3) << 1) + hh + ((v >> 2) << 4) + (khalf << 3);
}

__device__ __forceinline__ void wave_reduce2(float& s, float& q) {
#pragma unroll
  for (int off = 16; off > 0; off >>= 1) {
    s += __shfl_xor(s, off);
    q += __shfl_xor(q, off);
  }
}

// ---------------- K1: per-batch sum / sumsq of x (deterministic partials) ----------------
__global__ void __launch_bounds__(256)
k_stats_x(const float* __restrict__ x, float* __restrict__ part) {
  const int b = blockIdx.y;
  const float4* xb = (const float4*)(x + (size_t)b * kC * kP);
  const int tid = blockIdx.x * 256 + threadIdx.x;   // 65536 threads per batch
  float s = 0.f, q = 0.f;
#pragma unroll
  for (int i = 0; i < 16; ++i) {
    float4 v = xb[tid + i * 65536];
    s += (v.x + v.y) + (v.z + v.w);
    q += (v.x * v.x + v.y * v.y) + (v.z * v.z + v.w * v.w);
  }
  wave_reduce2(s, q);
  __shared__ float ls[8], lq[8];
  const int wave = threadIdx.x >> 5, lane = threadIdx.x & 31;
  if (lane == 0) { ls[wave] = s; lq[wave] = q; }
  __syncthreads();
  if (wave == 0) {
    s = (lane < 8) ? ls[lane] : 0.f;
    q = (lane < 8) ? lq[lane] : 0.f;
#pragma unroll
    for (int off = 4; off > 0; off >>= 1) { s += __shfl_xor(s, off); q += __shfl_xor(q, off); }
    if (lane == 0) {
      part[(b * kBlk1 + blockIdx.x) * 2 + 0] = s;
      part[(b * kBlk1 + blockIdx.x) * 2 + 1] = q;
    }
  }
}

// ---------------- deterministic partial -> final stats ----------------
__global__ void __launch_bounds__(256)
k_reduce_stats(const float* __restrict__ part, float* __restrict__ statsOut, int nblk) {
  const int b = blockIdx.x;
  float s = 0.f, q = 0.f;
  for (int i = threadIdx.x; i < nblk; i += 256) {
    s += part[(b * nblk + i) * 2 + 0];
    q += part[(b * nblk + i) * 2 + 1];
  }
  wave_reduce2(s, q);
  __shared__ float ls[8], lq[8];
  const int wave = threadIdx.x >> 5, lane = threadIdx.x & 31;
  if (lane == 0) { ls[wave] = s; lq[wave] = q; }
  __syncthreads();
  if (wave == 0) {
    s = (lane < 8) ? ls[lane] : 0.f;
    q = (lane < 8) ? lq[lane] : 0.f;
#pragma unroll
    for (int off = 4; off > 0; off >>= 1) { s += __shfl_xor(s, off); q += __shfl_xor(q, off); }
    if (lane == 0) { statsOut[b * 2] = s; statsOut[b * 2 + 1] = q; }
  }
}

// ---------------- K2: GN1(fold) + matmul1(WMMA) + SiLU -> stats of h ----------------
__global__ void __launch_bounds__(256)
k_mm1_stats(const float* __restrict__ x,
            const float* __restrict__ g1, const float* __restrict__ bta1,
            const float* __restrict__ w1, const float* __restrict__ b1,
            const float* __restrict__ stats, float* __restrict__ part) {
  __shared__ float w1p[kC * kC];
  __shared__ float sc1s[kC], sh1s[kC], b1eff[kC];
  __shared__ float ls[8], lq[8];

  const int b = blockIdx.y;
  const int tid = threadIdx.x;
  const float invN = 1.f / ((float)kC * (float)kP);
  const float mean1 = stats[b * 2] * invN;
  const float var1  = fmaxf(stats[b * 2 + 1] * invN - mean1 * mean1, 0.f);
  const float rs1   = rsqrtf(var1 + kEps);

  if (tid < kC) {
    float sc = rs1 * g1[tid];
    sc1s[tid] = sc;
    sh1s[tid] = bta1[tid] - mean1 * sc;
  }
  __syncthreads();
  // fold GN1 scale into weights; fold GN1 shift into an effective bias
  for (int i = tid; i < kC * kC; i += 256) w1p[i] = w1[i] * sc1s[i & (kC - 1)];
  if (tid < kC) {
    float acc = b1[tid];
    const float* wr = w1 + tid * kC;
#pragma unroll 4
    for (int k = 0; k < kC; ++k) acc += sh1s[k] * wr[k];
    b1eff[tid] = acc;
  }
  __syncthreads();

  const int lane = tid & 31, wave = tid >> 5;
  const int khalf = lane >> 4, mlane = lane & 15;

  // B fragments: B[k][n] = sc1[k]*w1[n][k]; lane%16 = n_local, element j = k_local (a_chan order)
  v16h bf[2][4];
#pragma unroll
  for (int ks = 0; ks < 2; ++ks)
#pragma unroll
    for (int t = 0; t < 4; ++t) {
      v16h bv;
#pragma unroll
      for (int j = 0; j < 16; ++j)
        bv[j] = (_Float16)w1p[(t * 16 + mlane) * kC + ks * 32 + a_chan(j, khalf)];
      bf[ks][t] = bv;
    }
  float myb1[4];
#pragma unroll
  for (int t = 0; t < 4; ++t) myb1[t] = b1eff[t * 16 + mlane];

  const float* xb = x + (size_t)b * kC * kP;
  const int tile0 = (blockIdx.x * 8 + wave) * kTPW;
  float s = 0.f, q = 0.f;

  for (int it = 0; it < kTPW; ++it) {
    const int p0 = (tile0 + it) * 16;
    const float* xp = xb + p0 + mlane;
    if (it + 1 < kTPW) __builtin_prefetch(xp + 16, 0, 0);
    v16h a[2];
#pragma unroll
    for (int ks = 0; ks < 2; ++ks) {
      v16h av;
#pragma unroll
      for (int j = 0; j < 16; ++j) {
        const int k = ks * 32 + a_chan(j, khalf);
        av[j] = (_Float16)xp[(size_t)k << 16];     // raw x -> f16 (GN folded into B)
      }
      a[ks] = av;
    }
    v8f acc[4];
    v8f vz = {};
#pragma unroll
    for (int t = 0; t < 4; ++t) acc[t] = vz;
#pragma unroll
    for (int ks = 0; ks < 2; ++ks)
#pragma unroll
      for (int t = 0; t < 4; ++t)
        acc[t] = __builtin_amdgcn_wmma_f32_16x16x32_f16(false, a[ks], false, bf[ks][t],
                                                        (short)0, acc[t], false, false);
#pragma unroll
    for (int t = 0; t < 4; ++t)
#pragma unroll
      for (int r = 0; r < 8; ++r) {
        float hv = acc[t][r] + myb1[t];
        float sl = hv / (1.f + __expf(-hv));       // SiLU
        s += sl; q += sl * sl;
      }
  }

  wave_reduce2(s, q);
  if (lane == 0) { ls[wave] = s; lq[wave] = q; }
  __syncthreads();
  if (wave == 0) {
    s = (lane < 8) ? ls[lane] : 0.f;
    q = (lane < 8) ? lq[lane] : 0.f;
#pragma unroll
    for (int off = 4; off > 0; off >>= 1) { s += __shfl_xor(s, off); q += __shfl_xor(q, off); }
    if (lane == 0) {
      part[(b * kBlk2 + blockIdx.x) * 2 + 0] = s;
      part[(b * kBlk2 + blockIdx.x) * 2 + 1] = q;
    }
  }
}

// ---------------- K3: recompute h, GN2(fold) + matmul2(WMMA) -> out ----------------
__global__ void __launch_bounds__(256)
k_mm2_out(const float* __restrict__ x,
          const float* __restrict__ g1, const float* __restrict__ bta1,
          const float* __restrict__ w1, const float* __restrict__ b1,
          const float* __restrict__ g2, const float* __restrict__ bta2,
          const float* __restrict__ w2, const float* __restrict__ b2,
          const float* __restrict__ stats, float* __restrict__ out) {
  __shared__ float w1p[kC * kC];
  __shared__ float sc1s[kC], sh1s[kC], b1eff[kC];
  __shared__ float sc2s[kC], sh2s[kC], b2eff[16];
  __shared__ unsigned int hbuf[8][16 * 36];   // per-wave f16 tile [m][c], row stride 36 u32 (72 halves, padded)
  __shared__ float outLds[8][kO * 17];        // per-wave [n][m], row stride 17 (bank-spread)

  const int b = blockIdx.y;
  const int tid = threadIdx.x;
  const float invN = 1.f / ((float)kC * (float)kP);

  const float mean1 = stats[b * 2] * invN;
  const float var1  = fmaxf(stats[b * 2 + 1] * invN - mean1 * mean1, 0.f);
  const float rs1   = rsqrtf(var1 + kEps);
  const float mean2 = stats[32 + b * 2] * invN;
  const float var2  = fmaxf(stats[32 + b * 2 + 1] * invN - mean2 * mean2, 0.f);
  const float rs2   = rsqrtf(var2 + kEps);

  if (tid < kC) {
    float sc = rs1 * g1[tid];
    sc1s[tid] = sc;
    sh1s[tid] = bta1[tid] - mean1 * sc;
    float sc2 = rs2 * g2[tid];
    sc2s[tid] = sc2;
    sh2s[tid] = bta2[tid] - mean2 * sc2;
  }
  __syncthreads();
  for (int i = tid; i < kC * kC; i += 256) w1p[i] = w1[i] * sc1s[i & (kC - 1)];
  if (tid < kC) {
    float acc = b1[tid];
    const float* wr = w1 + tid * kC;
#pragma unroll 4
    for (int k = 0; k < kC; ++k) acc += sh1s[k] * wr[k];
    b1eff[tid] = acc;
  }
  if (tid < 16) {
    float acc = 0.f;
    if (tid < kO) {
      acc = b2[tid];
      const float* wr = w2 + tid * kC;
      for (int c = 0; c < kC; ++c) acc += sh2s[c] * wr[c];
    }
    b2eff[tid] = acc;
  }
  __syncthreads();

  const int lane = tid & 31, wave = tid >> 5;
  const int khalf = lane >> 4, mlane = lane & 15;

  v16h bf[2][4];
#pragma unroll
  for (int ks = 0; ks < 2; ++ks)
#pragma unroll
    for (int t = 0; t < 4; ++t) {
      v16h bv;
#pragma unroll
      for (int j = 0; j < 16; ++j)
        bv[j] = (_Float16)w1p[(t * 16 + mlane) * kC + ks * 32 + a_chan(j, khalf)];
      bf[ks][t] = bv;
    }
  // B2[c][n] = sc2[c]*w2[n][c], n padded to 16 with zeros
  v16h b2f[2];
#pragma unroll
  for (int ks = 0; ks < 2; ++ks) {
    v16h bv;
#pragma unroll
    for (int j = 0; j < 16; ++j) {
      const int k = ks * 32 + a_chan(j, khalf);
      bv[j] = (mlane < kO) ? (_Float16)(sc2s[k] * w2[mlane * kC + k]) : (_Float16)0.f;
    }
    b2f[ks] = bv;
  }
  float myb1[4];
#pragma unroll
  for (int t = 0; t < 4; ++t) myb1[t] = b1eff[t * 16 + mlane];
  const float myb2 = b2eff[mlane];

  const float* xb = x + (size_t)b * kC * kP;
  float* outb = out + (size_t)b * kO * kP;
  const int tile0 = (blockIdx.x * 8 + wave) * kTPW;
  unsigned int* hrow = &hbuf[wave][0];
  float* orow = &outLds[wave][0];

  for (int it = 0; it < kTPW; ++it) {
    const int p0 = (tile0 + it) * 16;
    const float* xp = xb + p0 + mlane;
    if (it + 1 < kTPW) __builtin_prefetch(xp + 16, 0, 0);
    v16h a[2];
#pragma unroll
    for (int ks = 0; ks < 2; ++ks) {
      v16h av;
#pragma unroll
      for (int j = 0; j < 16; ++j) {
        const int k = ks * 32 + a_chan(j, khalf);
        av[j] = (_Float16)xp[(size_t)k << 16];
      }
      a[ks] = av;
    }
    v8f acc[4];
    v8f vz = {};
#pragma unroll
    for (int t = 0; t < 4; ++t) acc[t] = vz;
#pragma unroll
    for (int ks = 0; ks < 2; ++ks)
#pragma unroll
      for (int t = 0; t < 4; ++t)
        acc[t] = __builtin_amdgcn_wmma_f32_16x16x32_f16(false, a[ks], false, bf[ks][t],
                                                        (short)0, acc[t], false, false);
    // SiLU -> f16 (GN2 folded into B2/b2eff) -> LDS in [m][c] order (shuffle-pack b32 stores)
#pragma unroll
    for (int t = 0; t < 4; ++t)
#pragma unroll
      for (int r = 0; r < 8; ++r) {
        float hv = acc[t][r] + myb1[t];
        float sl = hv / (1.f + __expf(-hv));
        float hi = __shfl_xor(sl, 1);              // partner holds c^1
        if ((lane & 1) == 0) {                     // even lane owns even c pair
          const int c = t * 16 + mlane;            // even
          const int m = r + (khalf << 3);
          unsigned int u =
              (unsigned int)__builtin_bit_cast(unsigned short, (_Float16)sl) |
              ((unsigned int)__builtin_bit_cast(unsigned short, (_Float16)hi) << 16);
          hrow[m * 36 + (c >> 1)] = u;
        }
      }
    asm volatile("s_wait_dscnt 0" ::: "memory");   // cross-lane LDS visibility (same wave)

    // read back directly in A-fragment packing and run the 64->16(6) WMMA
    v8f acc2 = {};
#pragma unroll
    for (int ks = 0; ks < 2; ++ks) {
      v8u au;
#pragma unroll
      for (int vv = 0; vv < 8; ++vv) {
        const int cb = ks * 32 + a_chan(2 * vv, khalf);  // even base; pair (cb,cb+1) packed
        au[vv] = hrow[mlane * 36 + (cb >> 1)];
      }
      v16h a2 = __builtin_bit_cast(v16h, au);
      acc2 = __builtin_amdgcn_wmma_f32_16x16x32_f16(false, a2, false, b2f[ks],
                                                    (short)0, acc2, false, false);
    }
    // stage [n][m] in LDS, then coalesced 64B stores
    if (mlane < kO) {
#pragma unroll
      for (int r = 0; r < 8; ++r) {
        const int m = r + (khalf << 3);
        orow[mlane * 17 + m] = acc2[r] + myb2;
      }
    }
    asm volatile("s_wait_dscnt 0" ::: "memory");
    for (int i = lane; i < kO * 16; i += 32) {
      const int n = i >> 4, m = i & 15;
      outb[(size_t)n * kP + p0 + m] = orow[n * 17 + m];
    }
  }
}

// ---------------- host launch ----------------
extern "C" void kernel_launch(void* const* d_in, const int* in_sizes, int n_in,
                              void* d_out, int out_size, void* d_ws, size_t ws_size,
                              hipStream_t stream) {
  (void)in_sizes; (void)n_in; (void)out_size; (void)ws_size;
  const float* x    = (const float*)d_in[0];
  const float* g1   = (const float*)d_in[1];
  const float* bta1 = (const float*)d_in[2];
  const float* w1   = (const float*)d_in[3];
  const float* b1   = (const float*)d_in[4];
  const float* g2   = (const float*)d_in[5];
  const float* bta2 = (const float*)d_in[6];
  const float* w2   = (const float*)d_in[7];
  const float* b2   = (const float*)d_in[8];
  float* out = (float*)d_out;

  // ws layout (floats): [0..31] stats1, [32..63] stats2, then partials
  float* ws    = (float*)d_ws;
  float* stats = ws;
  float* part1 = ws + 64;                      // 16*256*2 = 8192 floats
  float* part2 = ws + 64 + kBatch * kBlk1 * 2; // 16*64*2  = 2048 floats

  k_stats_x<<<dim3(kBlk1, kBatch), 256, 0, stream>>>(x, part1);
  k_reduce_stats<<<kBatch, 256, 0, stream>>>(part1, stats, kBlk1);
  k_mm1_stats<<<dim3(kBlk2, kBatch), 256, 0, stream>>>(x, g1, bta1, w1, b1, stats, part2);
  k_reduce_stats<<<kBatch, 256, 0, stream>>>(part2, stats + 32, kBlk2);
  k_mm2_out<<<dim3(kBlk2, kBatch), 256, 0, stream>>>(x, g1, bta1, w1, b1, g2, bta2, w2, b2,
                                                     stats, out);
}
